// MyModel_70093866270860
// MI455X (gfx1250) — matrix operand order repaired and checked by reading, not compiled, not run on hardware
//
#include <hip/hip_runtime.h>

typedef _Float16 v16h __attribute__((ext_vector_type(16)));
typedef float    v8f  __attribute__((ext_vector_type(8)));

#define HID 10
#define NB  1024
#define NT  2048

union AF {
    v16h v;
    unsigned int u[8];
    _Float16 h[16];
};

__device__ __forceinline__ float fast_sigmoid(float x) {
    // 1 / (1 + exp(-x)) via v_exp_f32 (exp2) + v_rcp_f32
    float e = __builtin_amdgcn_exp2f(x * -1.44269504f);
    return __builtin_amdgcn_rcpf(1.0f + e);
}
__device__ __forceinline__ float fast_tanh(float x) {
    // tanh(x) = 2/(1+exp(-2x)) - 1
    float e = __builtin_amdgcn_exp2f(x * -2.88539008f);
    return __builtin_fmaf(2.0f, __builtin_amdgcn_rcpf(1.0f + e), -1.0f);
}

__launch_bounds__(32)
__global__ void lstm2_wmma_kernel(const float* __restrict__ x,
                                  const float* __restrict__ Wih0,
                                  const float* __restrict__ Whh0,
                                  const float* __restrict__ bih0,
                                  const float* __restrict__ bhh0,
                                  const float* __restrict__ Wih1,
                                  const float* __restrict__ Whh1,
                                  const float* __restrict__ bih1,
                                  const float* __restrict__ bhh1,
                                  const float* __restrict__ Wlin,
                                  const float* __restrict__ blin,
                                  float* __restrict__ out)
{
    // Per-wave staging rows: st[m][k] = A-matrix element (row m, k).
    // st0: layer0 A  = [x | h0(10) | 1 | 0...]
    // st1: layer1 A  = [h0(10) | x | 1 | h1(10) | 0...]  (also reused for linear A)
    __shared__ __align__(16) _Float16 st0[16][32];
    __shared__ __align__(16) _Float16 st1[16][32];

    const int tid = threadIdx.x;     // 0..31
    const int n   = tid & 15;        // B-col within tile / A-row m
    const int hi  = tid >> 4;        // lane half
    const int b0  = blockIdx.x * 16; // batch tile base

    // ---------------- Load B (weight) fragments once, f16 ----------------
    // B(k,n) layout (32x16 f16): lanes 0-15 hold K=0..15, lanes 16-31 K=16..31,
    // VGPR v holds the K pair (2v, 2v+1) as (lo,hi) halves.
    AF B0[4], B1[4], BL;
    for (int g = 0; g < 4; ++g) {              // gate: 0=i 1=f 2=g 3=o
        const int row = g * HID + n;           // row in the (40 x *) weight mats
        for (int e = 0; e < 16; ++e) {
            const int k = hi * 16 + e;
            float w0 = 0.f, w1 = 0.f;
            if (n < HID) {
                // layer0: K = [x, h0(10), bias]
                if (k == 0)            w0 = Wih0[row];                 // (40,1)
                else if (k <= 10)      w0 = Whh0[row * 10 + (k - 1)];  // (40,10)
                else if (k == 11)      w0 = bih0[row] + bhh0[row];
                // layer1: K = [h0(10), x, bias, h1(10)]
                if (k <= 9)            w1 = Wih1[row * 11 + k];        // (40,11)
                else if (k == 10)      w1 = Wih1[row * 11 + 10];
                else if (k == 11)      w1 = bih1[row] + bhh1[row];
                else if (k <= 21)      w1 = Whh1[row * 10 + (k - 12)]; // (40,10)
            }
            B0[g].h[e] = (_Float16)w0;
            B1[g].h[e] = (_Float16)w1;
        }
    }
    for (int e = 0; e < 16; ++e) {             // linear: only output col 0 valid
        const int k = hi * 16 + e;
        float w = 0.f;
        if (n == 0) {
            if (k == 10)                 w = Wlin[10];       // x coeff
            else if (k == 11)            w = blin[0];        // bias
            else if (k >= 12 && k <= 21) w = Wlin[k - 12];   // h1 coeffs
        }
        BL.h[e] = (_Float16)w;
    }

    // ---------------- Init staging (zeros + constant-1 bias column) -------
    for (int idx = tid; idx < 16 * 32; idx += 32) {
        ((_Float16*)st0)[idx] = (_Float16)0.f;
        ((_Float16*)st1)[idx] = (_Float16)0.f;
    }
    __syncthreads();
    if (tid < 16) {
        st0[tid][11] = (_Float16)1.f;
        st1[tid][11] = (_Float16)1.f;
    }
    __syncthreads();

    // Cell state in D-layout registers: c(m,j) at vgpr r=m%8, lane j+16*(m>=8)
    float c0[8], c1[8];
    #pragma unroll
    for (int r = 0; r < 8; ++r) { c0[r] = 0.f; c1[r] = 0.f; }

    const v8f zc = {0.f, 0.f, 0.f, 0.f, 0.f, 0.f, 0.f, 0.f};

    float xv = (tid < 16) ? x[(size_t)(b0 + tid) * NT] : 0.f;

    for (int t = 0; t < NT; ++t) {
        // publish x_t into staging
        if (tid < 16) {
            _Float16 xh = (_Float16)xv;
            st0[tid][0]  = xh;
            st1[tid][10] = xh;
        }
        // software prefetch x_{t+1}
        float xnext = 0.f;
        if (tid < 16 && t + 1 < NT) xnext = x[(size_t)(b0 + tid) * NT + t + 1];
        __syncthreads();

        // ---- A fragment load helper pattern:
        // A(m,k) f16 16x32: lane = m + 16*(k/16), vgpr = (k%16)>>1
        // => dword v of lane holds K pair starting at k = 16*hi + 2*v
        AF a;
        #pragma unroll
        for (int v = 0; v < 8; ++v) {
            const int k = 16 * hi + 2 * v;
            a.u[v] = *(const unsigned int*)&st0[n][k];
        }

        // ---- Layer 0: 4 gate WMMAs (bias folded via K=11 row) ----
        v8f gi = __builtin_amdgcn_wmma_f32_16x16x32_f16(false, a.v, false, B0[0].v, (short)0, zc, false, false);
        v8f gf = __builtin_amdgcn_wmma_f32_16x16x32_f16(false, a.v, false, B0[1].v, (short)0, zc, false, false);
        v8f gg = __builtin_amdgcn_wmma_f32_16x16x32_f16(false, a.v, false, B0[2].v, (short)0, zc, false, false);
        v8f go = __builtin_amdgcn_wmma_f32_16x16x32_f16(false, a.v, false, B0[3].v, (short)0, zc, false, false);

        _Float16 h0h[8];
        #pragma unroll
        for (int r = 0; r < 8; ++r) {
            float si = fast_sigmoid(gi[r]);
            float sf = fast_sigmoid(gf[r]);
            float so = fast_sigmoid(go[r]);
            float tg = fast_tanh(gg[r]);
            c0[r] = sf * c0[r] + si * tg;
            h0h[r] = (_Float16)(so * fast_tanh(c0[r]));
        }
        if (n < HID) {
            #pragma unroll
            for (int r = 0; r < 8; ++r) {
                const int m = 8 * hi + r;
                st0[m][1 + n] = h0h[r];   // for next step's layer0 A
                st1[m][n]     = h0h[r];   // for this step's layer1 A
            }
        }
        __syncthreads();

        // ---- Layer 1 ----
        #pragma unroll
        for (int v = 0; v < 8; ++v) {
            const int k = 16 * hi + 2 * v;
            a.u[v] = *(const unsigned int*)&st1[n][k];
        }
        gi = __builtin_amdgcn_wmma_f32_16x16x32_f16(false, a.v, false, B1[0].v, (short)0, zc, false, false);
        gf = __builtin_amdgcn_wmma_f32_16x16x32_f16(false, a.v, false, B1[1].v, (short)0, zc, false, false);
        gg = __builtin_amdgcn_wmma_f32_16x16x32_f16(false, a.v, false, B1[2].v, (short)0, zc, false, false);
        go = __builtin_amdgcn_wmma_f32_16x16x32_f16(false, a.v, false, B1[3].v, (short)0, zc, false, false);

        _Float16 h1h[8];
        #pragma unroll
        for (int r = 0; r < 8; ++r) {
            float si = fast_sigmoid(gi[r]);
            float sf = fast_sigmoid(gf[r]);
            float so = fast_sigmoid(go[r]);
            float tg = fast_tanh(gg[r]);
            c1[r] = sf * c1[r] + si * tg;
            h1h[r] = (_Float16)(so * fast_tanh(c1[r]));
        }
        if (n < HID) {
            #pragma unroll
            for (int r = 0; r < 8; ++r) {
                st1[8 * hi + r][12 + n] = h1h[r];   // h1_t replaces h1_{t-1}
            }
        }
        __syncthreads();

        // ---- Output linear: reuse st1 (h0 rows have zero weight in BL) ----
        #pragma unroll
        for (int v = 0; v < 8; ++v) {
            const int k = 16 * hi + 2 * v;
            a.u[v] = *(const unsigned int*)&st1[n][k];
        }
        v8f oacc = __builtin_amdgcn_wmma_f32_16x16x32_f16(false, a.v, false, BL.v, (short)0, zc, false, false);
        if (n == 0) {   // D col 0 lives in lanes 0 and 16; vgpr r -> row m = r + 8*hi
            #pragma unroll
            for (int r = 0; r < 8; ++r) {
                out[(size_t)(b0 + 8 * hi + r) * NT + t] = oacc[r];
            }
        }
        __syncthreads();   // keep next-iter staging stores behind this step's A reads
        xv = xnext;
    }
}

extern "C" void kernel_launch(void* const* d_in, const int* in_sizes, int n_in,
                              void* d_out, int out_size, void* d_ws, size_t ws_size,
                              hipStream_t stream) {
    (void)in_sizes; (void)n_in; (void)out_size; (void)d_ws; (void)ws_size;
    const float* x    = (const float*)d_in[0];
    const float* Wih0 = (const float*)d_in[1];
    const float* Whh0 = (const float*)d_in[2];
    const float* bih0 = (const float*)d_in[3];
    const float* bhh0 = (const float*)d_in[4];
    const float* Wih1 = (const float*)d_in[5];
    const float* Whh1 = (const float*)d_in[6];
    const float* bih1 = (const float*)d_in[7];
    const float* bhh1 = (const float*)d_in[8];
    const float* Wlin = (const float*)d_in[9];
    const float* blin = (const float*)d_in[10];
    // d_in[11] = future (0), ignored.
    float* out = (float*)d_out;

    lstm2_wmma_kernel<<<dim3(NB / 16), dim3(32), 0, stream>>>(
        x, Wih0, Whh0, bih0, bhh0, Wih1, Whh1, bih1, bhh1, Wlin, blin, out);
}
